// AdaptivePatching_14087492731171
// MI455X (gfx1250) — matrix operand order repaired and checked by reading, not compiled, not run on hardware
//
#include <hip/hip_runtime.h>
#include <hip/hip_bf16.h>
#include <math.h>

typedef __attribute__((ext_vector_type(16))) __bf16 v16bf;
typedef __attribute__((ext_vector_type(8)))  float  v8f;
typedef unsigned int u32x4 __attribute__((ext_vector_type(4)));
typedef int          i32x4 __attribute__((ext_vector_type(4)));
typedef int          i32x8 __attribute__((ext_vector_type(8)));

#define BB   32
#define CC   3
#define HH   128
#define WW   128
#define HID  64
#define NP   64
#define PP   16
#define EE   768
#define NHD  4
#define DH   192          // EE / NHD
#define TT   64           // tokens per batch (= channel count)
#define MTOK (BB * TT)    // 2048

// ---------------------------------------------------------------------------
// TDM prefetch: DMA a 64x16-f32 tile of the (N,K) weight tensor toward LDS
// scratch that nothing reads (pure L2 warm). Descriptor per ISA ch.8:
//   group0: count=1 | lds_addr | global_addr(57b) | type=2
//   group1: data_size=4B, tensor_dim0=K, tensor_dim1=64, tile=64x16, stride0=K
// All inputs are block-uniform -> SGPR descriptor groups.
// ---------------------------------------------------------------------------
__device__ __forceinline__ void tdm_prefetch_tile(const float* gptr, unsigned lds_off, int K)
{
    const unsigned long long ga = (unsigned long long)(uintptr_t)gptr;
    u32x4 g0;
    g0[0] = 1u;                                               // count=1, user desc
    g0[1] = lds_off;                                          // lds_addr (bytes)
    g0[2] = (unsigned)(ga & 0xFFFFFFFFu);                     // global_addr[31:0]
    g0[3] = (unsigned)((ga >> 32) & 0x01FFFFFFu) | 0x80000000u; // addr[56:32] | type=2
    i32x8 g1;
    g1[0] = 0x00020000;                                       // data_size=2 (4B), mask=0
    g1[1] = (int)(((unsigned)K & 0xFFFFu) << 16);             // tensor_dim0[15:0] @63:48
    g1[2] = (int)((((unsigned)K >> 16) & 0xFFFFu) | (64u << 16)); // dim0 hi | tensor_dim1=64
    g1[3] = (int)(64u << 16);                                 // tensor_dim1 hi=0 | tile_dim0=64
    g1[4] = 16;                                               // tile_dim1=16, tile_dim2=0
    g1[5] = K;                                                // tensor_dim0_stride[31:0]
    g1[6] = 0;
    g1[7] = 0;
    const i32x4 z4 = {0, 0, 0, 0};
#if __clang_major__ >= 23
    const i32x8 z8 = {0, 0, 0, 0, 0, 0, 0, 0};
    __builtin_amdgcn_tensor_load_to_lds(g0, g1, z4, z4, z8, 0);
#else
    __builtin_amdgcn_tensor_load_to_lds(g0, g1, z4, z4, 0);
#endif
}

// ---------------------------------------------------------------------------
// WMMA bf16 GEMM:  C[M,N] = act( A[M,K] @ W[N,K]^T + bias[N] )
// Block: 256 thr = 8 waves, 64(M) x 64(N) C tile; waves as 4(M) x 2(N), each
// wave owns 16x32 (two accumulators sharing one A fragment -> 2 WMMA/K-step).
// Per K-step(32): stage A(64x32 f32) and B(64x32 f32) through LDS as packed
// bf16 so each fragment is contiguous b128 ds_loads.
// grid = (N/64, M/64). Requires M%64==0, N%64==0, K%32==0 (all call sites do).
// ---------------------------------------------------------------------------
__global__ __launch_bounds__(256)
void gemm_bf16_wmma(const float* __restrict__ A, const float* __restrict__ Wt,
                    const float* __restrict__ bias, float* __restrict__ C,
                    int M, int N, int K, int relu)
{
    __shared__ __bf16 sA[64 * 32];        // [row][k]  row stride 32 bf16 (64 B)
    __shared__ __bf16 sB[64 * 32];        // [col][k]
    __shared__ float  tdmbuf[1024];       // TDM landing zone (never read)

    const int tid  = threadIdx.x;
    const int lane = tid & 31;
    const int wave = tid >> 5;
    const int mblk = blockIdx.y * 64;
    const int nblk = blockIdx.x * 64;

    // TDM prefetch of this block's first weight tile region (wave 0 only).
    if (wave == 0) {
        tdm_prefetch_tile(Wt + (size_t)nblk * K, (unsigned)(uintptr_t)tdmbuf, K);
    }

    const int l15 = lane & 15;
    const int lhi = lane >> 4;
    const int wm  = wave >> 1;            // 0..3
    const int wn  = wave & 1;             // 0..1
    const int m0  = mblk + wm * 16;
    const int n0  = nblk + wn * 32;

    // staging map: thread -> (row 0..63, 8-float segment 0..3)
    const int srow = tid >> 2;
    const int sseg = (tid & 3) * 8;
    const size_t agbase = (size_t)(mblk + srow) * K + sseg;
    const size_t bgbase = (size_t)(nblk + srow) * K + sseg;

    v8f acc0 = {}, acc1 = {};

    for (int k0 = 0; k0 < K; k0 += 32) {
        const float4 a0 = *reinterpret_cast<const float4*>(A  + agbase + k0);
        const float4 a1 = *reinterpret_cast<const float4*>(A  + agbase + k0 + 4);
        const float4 b0 = *reinterpret_cast<const float4*>(Wt + bgbase + k0);
        const float4 b1 = *reinterpret_cast<const float4*>(Wt + bgbase + k0 + 4);

        __syncthreads();                  // previous iteration fully consumed
        union { __bf16 h[8]; uint4 u; } pa, pb;
        pa.h[0] = (__bf16)a0.x; pa.h[1] = (__bf16)a0.y; pa.h[2] = (__bf16)a0.z; pa.h[3] = (__bf16)a0.w;
        pa.h[4] = (__bf16)a1.x; pa.h[5] = (__bf16)a1.y; pa.h[6] = (__bf16)a1.z; pa.h[7] = (__bf16)a1.w;
        pb.h[0] = (__bf16)b0.x; pb.h[1] = (__bf16)b0.y; pb.h[2] = (__bf16)b0.z; pb.h[3] = (__bf16)b0.w;
        pb.h[4] = (__bf16)b1.x; pb.h[5] = (__bf16)b1.y; pb.h[6] = (__bf16)b1.z; pb.h[7] = (__bf16)b1.w;
        *reinterpret_cast<uint4*>(&sA[srow * 32 + sseg]) = pa.u;
        *reinterpret_cast<uint4*>(&sB[srow * 32 + sseg]) = pb.u;
        __syncthreads();

        // A fragment: lane needs K {lhi*16..+15} contiguous
        v16bf af;
        {
            const __bf16* ar = &sA[(wm * 16 + l15) * 32];
            reinterpret_cast<uint4*>(&af)[0] = *reinterpret_cast<const uint4*>(ar + lhi * 16);
            reinterpret_cast<uint4*>(&af)[1] = *reinterpret_cast<const uint4*>(ar + lhi * 16 + 8);
        }
        // B fragments: lane needs K {lhi*16..+15} contiguous
        v16bf bf0, bf1;
        {
            const __bf16* br0 = &sB[(wn * 32 + l15) * 32 + lhi * 16];
            const __bf16* br1 = &sB[(wn * 32 + 16 + l15) * 32 + lhi * 16];
            reinterpret_cast<uint4*>(&bf0)[0] = *reinterpret_cast<const uint4*>(br0);
            reinterpret_cast<uint4*>(&bf0)[1] = *reinterpret_cast<const uint4*>(br0 + 8);
            reinterpret_cast<uint4*>(&bf1)[0] = *reinterpret_cast<const uint4*>(br1);
            reinterpret_cast<uint4*>(&bf1)[1] = *reinterpret_cast<const uint4*>(br1 + 8);
        }
        acc0 = __builtin_amdgcn_wmma_f32_16x16x32_bf16(false, af, false, bf0,
                                                       (short)0, acc0, false, false);
        acc1 = __builtin_amdgcn_wmma_f32_16x16x32_bf16(false, af, false, bf1,
                                                       (short)0, acc1, false, false);
    }

#pragma unroll
    for (int r = 0; r < 8; ++r) {
        const int mm = m0 + r + lhi * 8;
        const int nn = n0 + l15;
        float v0 = acc0[r] + bias[nn];
        float v1 = acc1[r] + bias[nn + 16];
        if (relu) { v0 = fmaxf(v0, 0.0f); v1 = fmaxf(v1, 0.0f); }
        C[(long)mm * N + nn]      = v0;
        C[(long)mm * N + nn + 16] = v1;
    }

    if (wave == 0) __builtin_amdgcn_s_wait_tensorcnt(0);
}

// ---------------------------------------------------------------------------
// Direct 3x3 SAME conv + bias + relu (small fraction of total FLOPs)
// ---------------------------------------------------------------------------
__global__ void conv3x3_relu_kernel(const float* __restrict__ in, const float* __restrict__ w,
                                    const float* __restrict__ bias, float* __restrict__ out,
                                    int Bn, int Cin, int Cout, int Hh, int Ww)
{
    long idx = (long)blockIdx.x * blockDim.x + threadIdx.x;
    long total = (long)Bn * Cout * Hh * Ww;
    if (idx >= total) return;
    int xx0 = (int)(idx % Ww); long t = idx / Ww;
    int yy0 = (int)(t % Hh);   t /= Hh;
    int co  = (int)(t % Cout);
    int b   = (int)(t / Cout);
    float acc = bias[co];
    for (int ci = 0; ci < Cin; ++ci) {
        const float* ip = in + (long)(b * Cin + ci) * Hh * Ww;
        const float* wp = w + (long)(co * Cin + ci) * 9;
#pragma unroll
        for (int dy = -1; dy <= 1; ++dy) {
            int yy = yy0 + dy;
            if (yy < 0 || yy >= Hh) continue;
#pragma unroll
            for (int dx = -1; dx <= 1; ++dx) {
                int xx = xx0 + dx;
                if (xx < 0 || xx >= Ww) continue;
                acc += ip[(long)yy * Ww + xx] * wp[(dy + 1) * 3 + (dx + 1)];
            }
        }
    }
    out[idx] = fmaxf(acc, 0.0f);
}

// ---------------------------------------------------------------------------
// 2x2 max pool, stride 2
// ---------------------------------------------------------------------------
__global__ void maxpool2_kernel(const float* __restrict__ in, float* __restrict__ out,
                                int planes, int Hh, int Ww)
{
    const int Ho = Hh >> 1, Wo = Ww >> 1;
    long idx = (long)blockIdx.x * blockDim.x + threadIdx.x;
    long total = (long)planes * Ho * Wo;
    if (idx >= total) return;
    int xo = (int)(idx % Wo); long t = idx / Wo;
    int yo = (int)(t % Ho);
    int pl = (int)(t / Ho);
    const float* ip = in + ((long)pl * Hh + 2 * yo) * Ww + 2 * xo;
    out[idx] = fmaxf(fmaxf(ip[0], ip[1]), fmaxf(ip[Ww], ip[Ww + 1]));
}

// ---------------------------------------------------------------------------
// MHA core: one block per (batch, head). T=64, dh=192. scores in LDS (16 KB).
// ---------------------------------------------------------------------------
__global__ __launch_bounds__(256)
void attn_kernel(const float* __restrict__ qkv, float* __restrict__ o)
{
    __shared__ float sc[TT * TT];
    const int b = blockIdx.x >> 2;
    const int h = blockIdx.x & 3;
    const float scale = 0.07216878364870323f;  // 1/sqrt(192)
    const int tid = threadIdx.x;

    for (int s = tid; s < TT * TT; s += 256) {
        const int qi = s >> 6, ki = s & 63;
        const float* qp = qkv + (long)(b * TT + qi) * (3 * EE) + h * DH;
        const float* kp = qkv + (long)(b * TT + ki) * (3 * EE) + EE + h * DH;
        float d = 0.0f;
        for (int e = 0; e < DH; ++e) d += qp[e] * kp[e];
        sc[s] = d * scale;
    }
    __syncthreads();
    if (tid < TT) {
        float* row = sc + tid * TT;
        float mx = row[0];
        for (int i = 1; i < TT; ++i) mx = fmaxf(mx, row[i]);
        float sum = 0.0f;
        for (int i = 0; i < TT; ++i) { float e = expf(row[i] - mx); row[i] = e; sum += e; }
        const float inv = 1.0f / sum;
        for (int i = 0; i < TT; ++i) row[i] *= inv;
    }
    __syncthreads();
    for (int s = tid; s < TT * DH; s += 256) {
        const int t = s / DH, d = s % DH;
        const float* vbase = qkv + (long)(b * TT) * (3 * EE) + 2 * EE + h * DH + d;
        const float* arow = sc + t * TT;
        float acc = 0.0f;
        for (int k = 0; k < TT; ++k) acc += arow[k] * vbase[(long)k * (3 * EE)];
        o[(long)(b * TT + t) * EE + h * DH + d] = acc;
    }
}

// ---------------------------------------------------------------------------
// fc2 (N=4) + affine-parameter math. One thread per token.
// ---------------------------------------------------------------------------
__global__ void fc2_params_kernel(const float* __restrict__ h1, const float* __restrict__ w,
                                  const float* __restrict__ bias, float* __restrict__ params,
                                  float* __restrict__ translate)
{
    const int m = blockIdx.x * blockDim.x + threadIdx.x;
    if (m >= MTOK) return;
    const float* hr = h1 + (long)m * 2048;
    float tp[4];
#pragma unroll
    for (int j = 0; j < 4; ++j) {
        const float* wr = w + (long)j * 2048;
        float acc = bias[j];
        for (int k = 0; k < 2048; ++k) acc += hr[k] * wr[k];
        tp[j] = acc;
    }
    float tx = tanhf(tp[0]);
    float ty = tanhf(tp[1]);
    const float s = (1.0f / (1.0f + expf(-tp[2]))) * 0.3f;   // sx == sy
    const float rot = tanhf(tp[3]) * 3.14159265358979323846f;
    const float c = cosf(rot), sn = sinf(rot);
    const float xe = fminf(s * fabsf(c) + s * fabsf(sn), 1.0f);
    const float ye = fminf(s * fabsf(sn) + s * fabsf(c), 1.0f);
    tx *= (1.0f - xe);
    ty *= (1.0f - ye);
    float* pp = params + m * 6;
    pp[0] = s * c;  pp[1] = -s * sn;
    pp[2] = s * sn; pp[3] = s * c;
    pp[4] = tx;     pp[5] = ty;
    translate[m * 2]     = tx;
    translate[m * 2 + 1] = ty;
}

// ---------------------------------------------------------------------------
// Bilinear grid sample of 16x16 patches from the original input image.
// ---------------------------------------------------------------------------
__global__ void grid_sample_kernel(const float* __restrict__ img, const float* __restrict__ params,
                                   float* __restrict__ patches)
{
    const int idx = blockIdx.x * blockDim.x + threadIdx.x;
    if (idx >= MTOK * PP * PP) return;
    const int px = idx & 15;
    const int py = (idx >> 4) & 15;
    const int m  = idx >> 8;
    const int b  = m >> 6;            // / NP
    const float* pp = params + m * 6;
    const float bx = (2.0f * px + 1.0f) / PP - 1.0f;
    const float by = (2.0f * py + 1.0f) / PP - 1.0f;
    const float gx = pp[0] * bx + pp[1] * by + pp[4];
    const float gy = pp[2] * bx + pp[3] * by + pp[5];
    const float xs = ((gx + 1.0f) * WW - 1.0f) * 0.5f;
    const float ys = ((gy + 1.0f) * HH - 1.0f) * 0.5f;
    const float x0f = floorf(xs), y0f = floorf(ys);
    const float wx = xs - x0f, wy = ys - y0f;
    const int x0 = (int)x0f, y0 = (int)y0f;
    const int x1 = x0 + 1,   y1 = y0 + 1;
#pragma unroll
    for (int c = 0; c < CC; ++c) {
        const float* ip = img + (long)(b * CC + c) * HH * WW;
        auto g = [&](int yy, int xx) -> float {
            const bool valid = (xx >= 0) && (xx < WW) && (yy >= 0) && (yy < HH);
            const int yc = min(max(yy, 0), HH - 1);
            const int xc = min(max(xx, 0), WW - 1);
            return valid ? ip[yc * WW + xc] : 0.0f;
        };
        const float v00 = g(y0, x0), v01 = g(y0, x1), v10 = g(y1, x0), v11 = g(y1, x1);
        const float out = v00 * (1 - wx) * (1 - wy) + v01 * wx * (1 - wy)
                        + v10 * (1 - wx) * wy       + v11 * wx * wy;
        patches[(long)(m * CC + c) * (PP * PP) + py * PP + px] = out;
    }
}

// ---------------------------------------------------------------------------
// Orchestration
// ---------------------------------------------------------------------------
extern "C" void kernel_launch(void* const* d_in, const int* in_sizes, int n_in,
                              void* d_out, int out_size, void* d_ws, size_t ws_size,
                              hipStream_t stream)
{
    const float* x        = (const float*)d_in[0];
    const float* conv1_w  = (const float*)d_in[1];
    const float* conv1_b  = (const float*)d_in[2];
    const float* a1_in_w  = (const float*)d_in[3];
    const float* a1_in_b  = (const float*)d_in[4];
    const float* a1_qkv_w = (const float*)d_in[5];
    const float* a1_qkv_b = (const float*)d_in[6];
    const float* a1_out_w = (const float*)d_in[7];
    const float* a1_out_b = (const float*)d_in[8];
    const float* a1_pj_w  = (const float*)d_in[9];
    const float* a1_pj_b  = (const float*)d_in[10];
    const float* conv2_w  = (const float*)d_in[11];
    const float* conv2_b  = (const float*)d_in[12];
    const float* a2_in_w  = (const float*)d_in[13];
    const float* a2_in_b  = (const float*)d_in[14];
    const float* a2_qkv_w = (const float*)d_in[15];
    const float* a2_qkv_b = (const float*)d_in[16];
    const float* a2_out_w = (const float*)d_in[17];
    const float* a2_out_b = (const float*)d_in[18];
    const float* a2_pj_w  = (const float*)d_in[19];
    const float* a2_pj_b  = (const float*)d_in[20];
    const float* fc1_w    = (const float*)d_in[21];
    const float* fc1_b    = (const float*)d_in[22];
    const float* fc2_w    = (const float*)d_in[23];
    const float* fc2_b    = (const float*)d_in[24];

    // workspace layout (floats); reuse: f1<->y1, pool<->y2, qkv<->h1
    float* ws   = (float*)d_ws;
    float* f1   = ws;                 // 33,554,432 : conv1 out, then proj1 out
    float* tok  = ws + 33554432;      //  1,572,864 : tokens
    float* qkv  = ws + 35127296;      //  4,718,592 : qkv, later h1
    float* att  = ws + 39845888;      //  1,572,864 : attention output
    float* tkb  = ws + 41418752;      //  1,572,864 : after out-proj
    float* pool = ws + 42991616;      //  8,388,608 : pooled, later y2
    float* f2   = ws + 51380224;      //  8,388,608 : conv2 out
    float* par  = ws + 59768832;      //     12,288 : affine params
    if (ws_size < (size_t)59781120 * sizeof(float)) return;

    float* patches   = (float*)d_out;
    float* translate = (float*)d_out + MTOK * CC * PP * PP;

    const dim3 blk(256);

    // Stage 1: conv1 + relu  (32,3,128,128) -> (32,64,128,128)
    conv3x3_relu_kernel<<<(BB * HID * HH * WW) / 256, blk, 0, stream>>>(
        x, conv1_w, conv1_b, f1, BB, CC, HID, HH, WW);

    // Attention block 1 (HW1 = 16384)
    gemm_bf16_wmma<<<dim3(EE / 64, MTOK / 64), blk, 0, stream>>>(
        f1, a1_in_w, a1_in_b, tok, MTOK, EE, 16384, 0);
    gemm_bf16_wmma<<<dim3(3 * EE / 64, MTOK / 64), blk, 0, stream>>>(
        tok, a1_qkv_w, a1_qkv_b, qkv, MTOK, 3 * EE, EE, 0);
    attn_kernel<<<BB * NHD, blk, 0, stream>>>(qkv, att);
    gemm_bf16_wmma<<<dim3(EE / 64, MTOK / 64), blk, 0, stream>>>(
        att, a1_out_w, a1_out_b, tkb, MTOK, EE, EE, 0);
    gemm_bf16_wmma<<<dim3(16384 / 64, MTOK / 64), blk, 0, stream>>>(
        tkb, a1_pj_w, a1_pj_b, f1, MTOK, 16384, EE, 0);   // y1 overwrites f1

    // maxpool + conv2 + relu
    maxpool2_kernel<<<(BB * HID * 64 * 64) / 256, blk, 0, stream>>>(f1, pool, BB * HID, HH, WW);
    conv3x3_relu_kernel<<<(BB * NP * 64 * 64) / 256, blk, 0, stream>>>(
        pool, conv2_w, conv2_b, f2, BB, HID, NP, 64, 64);

    // Attention block 2 (HW2 = 4096)
    gemm_bf16_wmma<<<dim3(EE / 64, MTOK / 64), blk, 0, stream>>>(
        f2, a2_in_w, a2_in_b, tok, MTOK, EE, 4096, 0);
    gemm_bf16_wmma<<<dim3(3 * EE / 64, MTOK / 64), blk, 0, stream>>>(
        tok, a2_qkv_w, a2_qkv_b, qkv, MTOK, 3 * EE, EE, 0);
    attn_kernel<<<BB * NHD, blk, 0, stream>>>(qkv, att);
    gemm_bf16_wmma<<<dim3(EE / 64, MTOK / 64), blk, 0, stream>>>(
        att, a2_out_w, a2_out_b, tkb, MTOK, EE, EE, 0);
    gemm_bf16_wmma<<<dim3(4096 / 64, MTOK / 64), blk, 0, stream>>>(
        tkb, a2_pj_w, a2_pj_b, pool, MTOK, 4096, EE, 0);  // y2 overwrites pool

    // fc1 (relu) + fc2/params + grid sample
    gemm_bf16_wmma<<<dim3(2048 / 64, MTOK / 64), blk, 0, stream>>>(
        pool, fc1_w, fc1_b, qkv, MTOK, 2048, 4096, 1);    // h1 overwrites qkv
    fc2_params_kernel<<<MTOK / 256, blk, 0, stream>>>(qkv, fc2_w, fc2_b, par, translate);
    grid_sample_kernel<<<(MTOK * PP * PP) / 256, blk, 0, stream>>>(x, par, patches);
}
